// CstNetS2_36369783062860
// MI455X (gfx1250) — compile-verified, matrix operand order checked by reading, not run on hardware
//
#include <hip/hip_runtime.h>
#include <math.h>

// ---------------------------------------------------------------------------
// CstNetS2 forward for gfx1250 (MI455X). All GEMMs go through
// v_wmma_f32_16x16x32_bf16 (f32 -> bf16 inputs, f32 accumulate).
// LDS tiles are stored pre-swizzled in WMMA fragment order so each lane's
// 16-element bf16 fragment is one contiguous 32B read (2x ds_load_b128).
// Interior tiles use an unguarded two-phase stage (batch loads, then stores).
// ---------------------------------------------------------------------------

#define BS    16
#define NPT   2048
#define C0    32
#define C1    64
#define C2    128
#define C3    256
#define NPRIM 4
#define NCLS  40
#define NNEI  32
#define S1    1024
#define S2    512

typedef __attribute__((ext_vector_type(16))) __bf16 v16bf;
typedef __attribute__((ext_vector_type(8)))  float  v8f;

// perm(k): position of K-index k inside a fragment-ordered 32-element row.
// half h's 16 fragment elements then live contiguously at [h*16, h*16+16):
//   slots 0..7  = k 0..7    slots 8..15  = k 16..23   (half 0)
//   slots 16..23= k 8..15   slots 24..31 = k 24..31   (half 1)
__device__ __forceinline__ int kperm(int k) {
  return (k & 7) | ((k & 8) << 1) | ((k & 16) >> 1);
}

#define LDS_STRIDE 48   // 32 used + 16 pad (bf16) -> 96B rows, 32B-aligned frags

// ---------------------------------------------------------------------------
// Generic GEMM: C[M,N] = act(A[M,K] @ W[K,N] + bias[N])
// Block = 128 threads (4 waves). Block tile 64(M) x 64(N), K-chunk 32.
// Wave w computes rows [w*16, w*16+16) x all 64 cols (4 WMMA per chunk).
// ---------------------------------------------------------------------------
__global__ __launch_bounds__(128) void k_gemm_wmma(
    const float* __restrict__ A, int lda,
    const float* __restrict__ W,
    const float* __restrict__ bias,
    float* __restrict__ C,
    int M, int N, int K, int relu)
{
  __shared__ __bf16 As[64 * LDS_STRIDE];   // A tile, fragment-order by row
  __shared__ __bf16 Bs[64 * LDS_STRIDE];   // W tile, fragment-order by column

  const int tid  = threadIdx.x;
  const int wave = tid >> 5;
  const int lane = tid & 31;
  const int half = lane >> 4;
  const int l16  = lane & 15;
  const int m0 = blockIdx.x * 64;
  const int n0 = blockIdx.y * 64;
  const bool interior = (m0 + 64 <= M) && (n0 + 64 <= N);

  // per-thread staging coordinates (constant across K chunks)
  const int ar_r[8]  = { (tid + 0*128) >> 4, (tid + 1*128) >> 4,
                         (tid + 2*128) >> 4, (tid + 3*128) >> 4,
                         (tid + 4*128) >> 4, (tid + 5*128) >> 4,
                         (tid + 6*128) >> 4, (tid + 7*128) >> 4 };
  const int ar_k = (tid & 15) << 1;           // same for all i (p&15 == tid&15)
  const int bc_c = tid & 63;                  // same for all i with 128 threads?
  // NOTE: for B, p = tid + i*128 -> c = p & 63 varies only via tid&63 when
  // i*128 % 64 == 0, which holds; k2 = ((p >> 6) << 1) = ((tid>>6) + 2*i)*2.
  const int bc_k0 = ((tid >> 6) << 1);        // 0 or 2

  v8f acc[4];
#pragma unroll
  for (int t = 0; t < 4; ++t)
#pragma unroll
    for (int e = 0; e < 8; ++e) acc[t][e] = 0.f;

  for (int k0 = 0; k0 < K; k0 += 32) {
    if (interior && (k0 + 32 <= K)) {
      // ---------- fast path: two-phase, no guards ----------
      float a0v[8], a1v[8];
#pragma unroll
      for (int i = 0; i < 8; ++i) {
        const float* ar = A + (size_t)(m0 + ar_r[i]) * lda + k0 + ar_k;
        a0v[i] = ar[0]; a1v[i] = ar[1];
      }
      float b0v[8], b1v[8];
#pragma unroll
      for (int i = 0; i < 8; ++i) {
        int k2 = bc_k0 + 4 * i;
        const float* wr = W + (size_t)(k0 + k2) * N + n0 + bc_c;
        b0v[i] = wr[0]; b1v[i] = wr[N];
      }
#pragma unroll
      for (int i = 0; i < 8; ++i) {
        union { __bf16 h[2]; unsigned u; } pk;
        pk.h[0] = (__bf16)a0v[i]; pk.h[1] = (__bf16)a1v[i];
        *(unsigned*)&As[ar_r[i] * LDS_STRIDE + kperm(ar_k)] = pk.u;
      }
#pragma unroll
      for (int i = 0; i < 8; ++i) {
        int k2 = bc_k0 + 4 * i;
        union { __bf16 h[2]; unsigned u; } pk;
        pk.h[0] = (__bf16)b0v[i]; pk.h[1] = (__bf16)b1v[i];
        *(unsigned*)&Bs[bc_c * LDS_STRIDE + kperm(k2)] = pk.u;
      }
    } else {
      // ---------- guarded path: clamped addresses + selects ----------
#pragma unroll
      for (int i = 0; i < 8; ++i) {
        int p  = tid + i * 128;
        int r  = p >> 4;
        int k2 = (p & 15) << 1;
        int gr = m0 + r;
        int kk = k0 + k2;
        int grc = gr < M ? gr : M - 1;
        int kc0 = kk < K ? kk : 0;
        int kc1 = (kk + 1) < K ? (kk + 1) : 0;
        const float* ar = A + (size_t)grc * lda;
        float a0 = ar[kc0];
        float a1 = ar[kc1];
        a0 = (gr < M && kk     < K) ? a0 : 0.f;
        a1 = (gr < M && kk + 1 < K) ? a1 : 0.f;
        union { __bf16 h[2]; unsigned u; } pk;
        pk.h[0] = (__bf16)a0; pk.h[1] = (__bf16)a1;
        *(unsigned*)&As[r * LDS_STRIDE + kperm(k2)] = pk.u;
      }
#pragma unroll
      for (int i = 0; i < 8; ++i) {
        int p  = tid + i * 128;
        int c  = p & 63;
        int k2 = (p >> 6) << 1;
        int gc = n0 + c, gk = k0 + k2;
        int gcc = gc < N ? gc : N - 1;
        int gk0 = gk < K ? gk : 0;
        int gk1 = (gk + 1) < K ? (gk + 1) : 0;
        float b0 = W[(size_t)gk0 * N + gcc];
        float b1 = W[(size_t)gk1 * N + gcc];
        b0 = (gc < N && gk     < K) ? b0 : 0.f;
        b1 = (gc < N && gk + 1 < K) ? b1 : 0.f;
        union { __bf16 h[2]; unsigned u; } pk;
        pk.h[0] = (__bf16)b0; pk.h[1] = (__bf16)b1;
        *(unsigned*)&Bs[c * LDS_STRIDE + kperm(k2)] = pk.u;
      }
    }
    // prefetch next A tile (global_prefetch_b8)
    if (k0 + 32 < K) {
      int gr = m0 + (tid >> 1);
      if (gr < M) __builtin_prefetch(&A[(size_t)gr * lda + k0 + 32], 0, 0);
    }
    __syncthreads();

    // fragments: contiguous 32B per lane -> 2x ds_load_b128 each
    const v16bf afrag =
        *(const v16bf*)&As[(wave * 16 + l16) * LDS_STRIDE + half * 16];
#pragma unroll
    for (int t = 0; t < 4; ++t) {
      const v16bf bfrag =
          *(const v16bf*)&Bs[(t * 16 + l16) * LDS_STRIDE + half * 16];
      acc[t] = __builtin_amdgcn_wmma_f32_16x16x32_bf16(
          false, afrag, false, bfrag, (short)0, acc[t], false, false);
    }
    __syncthreads();
  }

  // epilogue: D layout — VGPR r: row = r + half*8, col = l16 (+ subtile)
#pragma unroll
  for (int t = 0; t < 4; ++t) {
    int gc = n0 + t * 16 + l16;
    if (gc >= N) continue;
    float bv = bias ? bias[gc] : 0.f;
#pragma unroll
    for (int r = 0; r < 8; ++r) {
      int gr = m0 + wave * 16 + r + half * 8;
      if (gr < M) {
        float v = acc[t][r] + bv;
        if (relu) v = v > 0.f ? v : 0.f;
        C[(size_t)gr * N + gc] = v;
      }
    }
  }
}

// ---------------------------------------------------------------------------
// KNN: per (b,i) compute all n distances into LDS, then 32x argmin-select.
// grid = (n, BS), block = 256
// ---------------------------------------------------------------------------
__global__ __launch_bounds__(256) void k_knn(
    const float* __restrict__ xyz, int n, int* __restrict__ idx)
{
  __shared__ float dist[2048];
  __shared__ float rmin[256];
  __shared__ int   rarg[256];
  const int b = blockIdx.y, i = blockIdx.x, t = threadIdx.x;
  const float* xb = xyz + (size_t)b * n * 3;
  const float xi = xb[i * 3], yi = xb[i * 3 + 1], zi = xb[i * 3 + 2];
  for (int j = t; j < n; j += 256) {
    float dx = xb[j * 3] - xi, dy = xb[j * 3 + 1] - yi, dz = xb[j * 3 + 2] - zi;
    dist[j] = dx * dx + dy * dy + dz * dz;
  }
  __syncthreads();
  for (int kk = 0; kk < NNEI; ++kk) {
    float best = 3.0e38f; int barg = 0;
    for (int j = t; j < n; j += 256) {
      float d = dist[j];
      if (d < best) { best = d; barg = j; }
    }
    rmin[t] = best; rarg[t] = barg;
    __syncthreads();
    for (int s = 128; s > 0; s >>= 1) {
      if (t < s) {
        if (rmin[t + s] < rmin[t] ||
            (rmin[t + s] == rmin[t] && rarg[t + s] < rarg[t])) {
          rmin[t] = rmin[t + s]; rarg[t] = rarg[t + s];
        }
      }
      __syncthreads();
    }
    if (t == 0) {
      idx[((size_t)b * n + i) * NNEI + kk] = rarg[0];
      dist[rarg[0]] = 3.0e38f;
    }
    __syncthreads();
  }
}

// ---------------------------------------------------------------------------
// Farthest point sampling. grid = BS, block = 256.
// Matches jax scan: record current farthest, then update dists + argmax.
// ---------------------------------------------------------------------------
__global__ __launch_bounds__(256) void k_fps(
    const float* __restrict__ xyz, int n, int ncenter, int* __restrict__ out)
{
  __shared__ float dist[2048];
  __shared__ float rmax[256];
  __shared__ int   rarg[256];
  __shared__ int   far_s;
  const int b = blockIdx.x, t = threadIdx.x;
  const float* xb = xyz + (size_t)b * n * 3;
  for (int j = t; j < n; j += 256) dist[j] = 1e10f;
  if (t == 0) far_s = 0;
  __syncthreads();
  for (int it = 0; it < ncenter; ++it) {
    int f = far_s;
    if (t == 0) out[(size_t)b * ncenter + it] = f;
    float cx = xb[f * 3], cy = xb[f * 3 + 1], cz = xb[f * 3 + 2];
    float best = -1.f; int barg = 0;
    for (int j = t; j < n; j += 256) {
      float dx = xb[j * 3] - cx, dy = xb[j * 3 + 1] - cy, dz = xb[j * 3 + 2] - cz;
      float d = dx * dx + dy * dy + dz * dz;
      float nd = dist[j]; nd = nd < d ? nd : d; dist[j] = nd;
      if (nd > best) { best = nd; barg = j; }
    }
    rmax[t] = best; rarg[t] = barg;
    __syncthreads();
    for (int s = 128; s > 0; s >>= 1) {
      if (t < s) {
        if (rmax[t + s] > rmax[t] ||
            (rmax[t + s] == rmax[t] && rarg[t + s] < rarg[t])) {
          rmax[t] = rmax[t + s]; rarg[t] = rarg[t + s];
        }
      }
      __syncthreads();
    }
    if (t == 0) far_s = rarg[0];
    __syncthreads();
  }
}

// out[b,s,c] = in[b, idx[b,s], c]
__global__ void k_gather_f32(const float* __restrict__ in, const int* __restrict__ idx,
                             float* __restrict__ out, int n_in, int S, int C)
{
  size_t t = (size_t)blockIdx.x * blockDim.x + threadIdx.x;
  size_t total = (size_t)BS * S * C;
  if (t >= total) return;
  int c = t % C; size_t bs = t / C; int s = bs % S; int b = bs / S;
  int j = idx[(size_t)b * S + s];
  out[t] = in[((size_t)b * n_in + j) * C + c];
}

// out[b,s,k] = in[(b, fidx[b,s], k)]
__global__ void k_gather_rows(const int* __restrict__ in, const int* __restrict__ fidx,
                              int* __restrict__ out, int n_in, int S)
{
  size_t t = (size_t)blockIdx.x * blockDim.x + threadIdx.x;
  size_t total = (size_t)BS * S * NNEI;
  if (t >= total) return;
  int k = t % NNEI; size_t bs = t / NNEI; int s = bs % S; int b = bs / S;
  int j = fidx[(size_t)b * S + s];
  out[t] = in[((size_t)b * n_in + j) * NNEI + k];
}

// Build grouped feature rows [B,S,K,cg].
// mode 0: [fea(Cf) | raw_g - raw_c]     (cg = Cf + D)
// mode 1: [fea(Cf) | raw_g | raw_c]     (cg = Cf + 2D)
__global__ void k_build_gfea(const float* __restrict__ fea, int Cf,
                             const float* __restrict__ raw, int D,
                             const float* __restrict__ craw,
                             const int* __restrict__ idx,
                             int n_in, int S, float* __restrict__ out, int mode)
{
  size_t t = (size_t)blockIdx.x * blockDim.x + threadIdx.x;
  size_t total = (size_t)BS * S * NNEI;
  if (t >= total) return;
  size_t bs = t / NNEI; int s = bs % S; int b = bs / S;
  int j = idx[t];
  const float* fr = fea + ((size_t)b * n_in + j) * Cf;
  const float* rr = raw + ((size_t)b * n_in + j) * D;
  const float* rc = craw + ((size_t)b * S + s) * D;
  int cg = Cf + (mode ? 2 * D : D);
  float* o = out + t * cg;
  for (int c = 0; c < Cf; ++c) o[c] = fr[c];
  if (mode == 0) {
    for (int d = 0; d < D; ++d) o[Cf + d] = rr[d] - rc[d];
  } else {
    for (int d = 0; d < D; ++d) { o[Cf + d] = rr[d]; o[Cf + D + d] = rc[d]; }
  }
}

// Point attention: one wave per (b,s). softmax over 32 neighbors via shfl.
__global__ __launch_bounds__(32) void k_point_attn(
    const float* __restrict__ q,    // [R, co]
    const float* __restrict__ kmat, // [R, 32, co]
    const float* __restrict__ vmat, // [R, 32, co]
    float* __restrict__ out,        // [R, co]
    int co, float scale)
{
  __shared__ float q_sh[256];
  __shared__ float a_sh[NNEI];
  const int row = blockIdx.x, lane = threadIdx.x;
  const float* qr = q + (size_t)row * co;
  for (int c = lane; c < co; c += 32) q_sh[c] = qr[c];
  __syncthreads();
  const float* kr = kmat + ((size_t)row * NNEI + lane) * co;
  float s = 0.f;
  for (int c = 0; c < co; ++c) s += q_sh[c] * kr[c];
  s *= scale;
  float m = s;
  for (int off = 16; off; off >>= 1) m = fmaxf(m, __shfl_xor(m, off));
  float e = __expf(s - m);
  float sum = e;
  for (int off = 16; off; off >>= 1) sum += __shfl_xor(sum, off);
  a_sh[lane] = e / sum;
  __syncthreads();
  const float* vr = vmat + (size_t)row * NNEI * co;
  for (int c = lane; c < co; c += 32) {
    float acc = 0.f;
    for (int k = 0; k < NNEI; ++k) acc += a_sh[k] * vr[(size_t)k * co + c];
    out[(size_t)row * co + c] = acc;
  }
}

// fea_attention fuse: softmax over 4 streams, weighted sum. One wave per row.
__global__ __launch_bounds__(32) void k_fea_fuse(
    const float* __restrict__ f0, const float* __restrict__ f1,
    const float* __restrict__ f2, const float* __restrict__ f3,
    const float* __restrict__ wW, const float* __restrict__ wb,
    float* __restrict__ fused, int C)
{
  const int row = blockIdx.x, lane = threadIdx.x;
  const float* fs0 = f0 + (size_t)row * C;
  const float* fs1 = f1 + (size_t)row * C;
  const float* fs2 = f2 + (size_t)row * C;
  const float* fs3 = f3 + (size_t)row * C;
  const float* fsa[4] = {fs0, fs1, fs2, fs3};
  float s[4];
#pragma unroll
  for (int n = 0; n < 4; ++n) {
    float p = 0.f;
    for (int c = lane; c < C; c += 32) p += fsa[n][c] * wW[c];
    for (int off = 16; off; off >>= 1) p += __shfl_xor(p, off);
    s[n] = p + wb[0];
  }
  float m = fmaxf(fmaxf(s[0], s[1]), fmaxf(s[2], s[3]));
  float e0 = __expf(s[0] - m), e1 = __expf(s[1] - m);
  float e2 = __expf(s[2] - m), e3 = __expf(s[3] - m);
  float inv = 1.f / (e0 + e1 + e2 + e3);
  float a0 = e0 * inv, a1 = e1 * inv, a2 = e2 * inv, a3 = e3 * inv;
  for (int c = lane; c < C; c += 32)
    fused[(size_t)row * C + c] = a0 * fs0[c] + a1 * fs1[c] + a2 * fs2[c] + a3 * fs3[c];
}

// cst stem input concat [xyz|mad|adj|pt] -> [B*N, 12]
__global__ void k_concat_cst(const float* __restrict__ xyz, const float* __restrict__ mad,
                             const float* __restrict__ adj, const float* __restrict__ pt,
                             float* __restrict__ out)
{
  size_t t = (size_t)blockIdx.x * blockDim.x + threadIdx.x;
  if (t >= (size_t)BS * NPT) return;
  float* o = out + t * 12;
  const float* x = xyz + t * 3;
  const float* m = mad + t * 3;
  const float* a = adj + t * 2;
  const float* p = pt + t * 4;
  o[0] = x[0]; o[1] = x[1]; o[2] = x[2];
  o[3] = m[0]; o[4] = m[1]; o[5] = m[2];
  o[6] = a[0]; o[7] = a[1];
  o[8] = p[0]; o[9] = p[1]; o[10] = p[2]; o[11] = p[3];
}

// global max over points of channel-concat [fea(Cf) | tile(raw, tile)]
__global__ void k_global_maxcat(const float* __restrict__ fea, int Cf,
                                const float* __restrict__ raw, int D, int tile,
                                int n, float* __restrict__ out)
{
  int Cc = Cf + D * tile;
  int t = blockIdx.x * blockDim.x + threadIdx.x;
  if (t >= BS * Cc) return;
  int b = t / Cc, c = t % Cc;
  float m = -3.4e38f;
  if (c < Cf) {
    const float* p = fea + (size_t)b * n * Cf + c;
    for (int j = 0; j < n; ++j) m = fmaxf(m, p[(size_t)j * Cf]);
  } else {
    int d = (c - Cf) % D;
    const float* p = raw + (size_t)b * n * D + d;
    for (int j = 0; j < n; ++j) m = fmaxf(m, p[(size_t)j * D]);
  }
  out[t] = m;
}

__global__ __launch_bounds__(64) void k_logsoftmax(const float* __restrict__ in,
                                                   float* __restrict__ out, int N)
{
  __shared__ float red[64];
  const int b = blockIdx.x, t = threadIdx.x;
  float v = (t < N) ? in[b * N + t] : -3.4e38f;
  red[t] = v; __syncthreads();
  for (int s = 32; s; s >>= 1) { if (t < s) red[t] = fmaxf(red[t], red[t + s]); __syncthreads(); }
  float m = red[0]; __syncthreads();
  float e = (t < N) ? __expf(v - m) : 0.f;
  red[t] = e; __syncthreads();
  for (int s = 32; s; s >>= 1) { if (t < s) red[t] += red[t + s]; __syncthreads(); }
  float lse = logf(red[0]) + m;
  if (t < N) out[b * N + t] = v - lse;
}

// ---------------------------------------------------------------------------
// Host orchestration.
// Param leaf order assumes jax pytree flatten (dict keys sorted) appended
// after the 4 data inputs:
//   4-7 adj0 | 8-11 cst0 | 12-15 fa1(p.W,p.b,w.W,w.b) | 16-19 fa2 | 20-23 fa3
//   24-27 head | 28-31 mad0 | 32-35 pt0
//   36-59 ssa1 (attn_adj, attn_cst, attn_mad, attn_pt; each k.W,k.b,q.W,q.b,v.W,v.b)
//   60-83 ssa2 (same) | 84-99 ssa3 (mlp_adj, mlp_cst, mlp_mad, mlp_pt; 2 lins each)
// ---------------------------------------------------------------------------
extern "C" void kernel_launch(void* const* d_in, const int* in_sizes, int n_in,
                              void* d_out, int out_size, void* d_ws, size_t ws_size,
                              hipStream_t stream)
{
  (void)in_sizes; (void)n_in; (void)out_size; (void)ws_size;
  const float* xyz = (const float*)d_in[0];
  const float* mad = (const float*)d_in[1];
  const float* adj = (const float*)d_in[2];
  const float* pt  = (const float*)d_in[3];
  auto Pw = [&](int i) { return (const float*)d_in[i]; };

  char* base = (char*)d_ws;
  size_t off = 0;
  auto allocB = [&](size_t bytes) -> void* {
    void* p = base + off;
    off += bytes; off = (off + 255) & ~(size_t)255;
    return p;
  };
  auto af = [&](size_t nelem) { return (float*)allocB(nelem * sizeof(float)); };
  auto ai = [&](size_t nelem) { return (int*)allocB(nelem * sizeof(int)); };

  auto gemm = [&](const float* A, int lda, const float* W, const float* b,
                  float* C, int M, int N, int K, int relu) {
    dim3 g((M + 63) / 64, (N + 63) / 64);
    k_gemm_wmma<<<g, 128, 0, stream>>>(A, lda, W, b, C, M, N, K, relu);
  };

  // ---- shared big scratch (reused across streams/levels) ----
  float* gbuf = af((size_t)BS * S1 * NNEI * 40);   // grouped features (max cg=40)
  float* kbuf = af((size_t)BS * S1 * NNEI * C1);   // == BS*S2*NNEI*C2
  float* vbuf = af((size_t)BS * S1 * NNEI * C1);
  float* qbuf = af((size_t)BS * S1 * C1);          // == BS*S2*C2

  // ---- stems ----
  const int M0 = BS * NPT;
  float* tmp16 = af((size_t)M0 * 16);
  float* madF  = af((size_t)M0 * C0);
  float* adjF  = af((size_t)M0 * C0);
  float* ptF   = af((size_t)M0 * C0);
  float* cstF  = af((size_t)M0 * C0);
  float* cstin = af((size_t)M0 * 12);

  gemm(mad, 3, Pw(28), Pw(29), tmp16, M0, 16, 3, 1);
  gemm(tmp16, 16, Pw(30), Pw(31), madF, M0, C0, 16, 1);
  gemm(adj, 2, Pw(4), Pw(5), tmp16, M0, 16, 2, 1);
  gemm(tmp16, 16, Pw(6), Pw(7), adjF, M0, C0, 16, 1);
  gemm(pt, 4, Pw(32), Pw(33), tmp16, M0, 16, 4, 1);
  gemm(tmp16, 16, Pw(34), Pw(35), ptF, M0, C0, 16, 1);
  k_concat_cst<<<(M0 + 255) / 256, 256, 0, stream>>>(xyz, mad, adj, pt, cstin);
  gemm(cstin, 12, Pw(8), Pw(9), tmp16, M0, 16, 12, 1);
  gemm(tmp16, 16, Pw(10), Pw(11), cstF, M0, C0, 16, 1);

  // ---- one quartic SSA level (streams in order: mad, adj, pt, cst) ----
  auto run_level = [&](int n, int S, int Cin, int Cout,
                       const float* xl, const float* ml, const float* al, const float* pl,
                       const float* feas[4], const int* bases,
                       float** ox, float** om, float** oa, float** op, float* outs[4]) {
    int* idx_all = ai((size_t)BS * n * NNEI);
    k_knn<<<dim3(n, BS), 256, 0, stream>>>(xl, n, idx_all);
    int* fidx = ai((size_t)BS * S);
    k_fps<<<BS, 256, 0, stream>>>(xl, n, S, fidx);

    auto gat = [&](const float* in, int C) {
      float* o = af((size_t)BS * S * C);
      size_t tot = (size_t)BS * S * C;
      k_gather_f32<<<(tot + 255) / 256, 256, 0, stream>>>(in, fidx, o, n, S, C);
      return o;
    };
    float* cx = gat(xl, 3);
    float* cm = gat(ml, 3);
    float* ca = gat(al, 2);
    float* cp = gat(pl, NPRIM);
    float* cfea[4];
    for (int st = 0; st < 4; ++st) cfea[st] = gat(feas[st], Cin);

    int* idx2 = ai((size_t)BS * S * NNEI);
    {
      size_t tot = (size_t)BS * S * NNEI;
      k_gather_rows<<<(tot + 255) / 256, 256, 0, stream>>>(idx_all, fidx, idx2, n, S);
    }

    const float* raws[4]  = {ml, al, pl, xl};
    const float* craws[4] = {cm, ca, cp, cx};
    const int Ds[4] = {3, 2, NPRIM, 3};
    const int modes[4] = {0, 1, 1, 0};
    const float scale = 1.0f / sqrtf((float)Cout);

    for (int st = 0; st < 4; ++st) {
      int cg = Cin + (modes[st] ? 2 * Ds[st] : Ds[st]);
      size_t rows = (size_t)BS * S * NNEI;
      k_build_gfea<<<(rows + 255) / 256, 256, 0, stream>>>(
          feas[st], Cin, raws[st], Ds[st], craws[st], idx2, n, S, gbuf, modes[st]);
      int bI = bases[st];
      gemm(cfea[st], Cin, Pw(bI + 2), Pw(bI + 3), qbuf, BS * S, Cout, Cin, 0);   // q
      gemm(gbuf, cg, Pw(bI + 0), Pw(bI + 1), kbuf, (int)rows, Cout, cg, 0);      // k
      gemm(gbuf, cg, Pw(bI + 4), Pw(bI + 5), vbuf, (int)rows, Cout, cg, 0);      // v
      outs[st] = af((size_t)BS * S * Cout);
      k_point_attn<<<BS * S, 32, 0, stream>>>(qbuf, kbuf, vbuf, outs[st], Cout, scale);
    }
    *ox = cx; *om = cm; *oa = ca; *op = cp;
  };

  // ---- level 1: 2048 -> 1024, C0 -> C1 ----
  const float* feas1[4] = {madF, adjF, ptF, cstF};
  const int bases1[4] = {48, 36, 54, 42};   // mad, adj, pt, cst
  float *x1, *m1, *a1, *p1; float* o1[4];
  run_level(NPT, S1, C0, C1, xyz, mad, adj, pt, feas1, bases1, &x1, &m1, &a1, &p1, o1);
  float* fused1 = af((size_t)BS * S1 * C1);
  k_fea_fuse<<<BS * S1, 32, 0, stream>>>(o1[0], o1[1], o1[2], o1[3], Pw(14), Pw(15), fused1, C1);
  float* cf1 = af((size_t)BS * S1 * C1);
  gemm(fused1, C1, Pw(12), Pw(13), cf1, BS * S1, C1, C1, 1);

  // ---- level 2: 1024 -> 512, C1 -> C2 ----
  const float* feas2[4] = {o1[0], o1[1], o1[2], cf1};
  const int bases2[4] = {72, 60, 78, 66};
  float *x2, *m2, *a2, *p2; float* o2[4];
  run_level(S1, S2, C1, C2, x1, m1, a1, p1, feas2, bases2, &x2, &m2, &a2, &p2, o2);
  float* fused2 = af((size_t)BS * S2 * C2);
  k_fea_fuse<<<BS * S2, 32, 0, stream>>>(o2[0], o2[1], o2[2], o2[3], Pw(18), Pw(19), fused2, C2);
  float* cf2 = af((size_t)BS * S2 * C2);
  gemm(fused2, C2, Pw(16), Pw(17), cf2, BS * S2, C2, C2, 1);

  // ---- level 3: global pooling + MLPs ----
  float* gm_in = af((size_t)BS * (C2 + 3));
  float* ga_in = af((size_t)BS * (C2 + 4));
  float* gp_in = af((size_t)BS * (C2 + 8));
  float* gc_in = af((size_t)BS * (C2 + 3));
  {
    int tot;
    tot = BS * (C2 + 3);
    k_global_maxcat<<<(tot + 127) / 128, 128, 0, stream>>>(o2[0], C2, m2, 3, 1, S2, gm_in);
    tot = BS * (C2 + 4);
    k_global_maxcat<<<(tot + 127) / 128, 128, 0, stream>>>(o2[1], C2, a2, 2, 2, S2, ga_in);
    tot = BS * (C2 + 8);
    k_global_maxcat<<<(tot + 127) / 128, 128, 0, stream>>>(o2[2], C2, p2, NPRIM, 2, S2, gp_in);
    tot = BS * (C2 + 3);
    k_global_maxcat<<<(tot + 127) / 128, 128, 0, stream>>>(cf2, C2, x2, 3, 1, S2, gc_in);
  }
  float* t3 = af((size_t)BS * C3);
  float* gm = af((size_t)BS * C3);
  float* ga = af((size_t)BS * C3);
  float* gp = af((size_t)BS * C3);
  float* gc = af((size_t)BS * C3);
  gemm(gm_in, C2 + 3, Pw(92), Pw(93), t3, BS, C3, C2 + 3, 1);   // mlp_mad
  gemm(t3, C3, Pw(94), Pw(95), gm, BS, C3, C3, 1);
  gemm(ga_in, C2 + 4, Pw(84), Pw(85), t3, BS, C3, C2 + 4, 1);   // mlp_adj
  gemm(t3, C3, Pw(86), Pw(87), ga, BS, C3, C3, 1);
  gemm(gp_in, C2 + 8, Pw(96), Pw(97), t3, BS, C3, C2 + 8, 1);   // mlp_pt
  gemm(t3, C3, Pw(98), Pw(99), gp, BS, C3, C3, 1);
  gemm(gc_in, C2 + 3, Pw(88), Pw(89), t3, BS, C3, C2 + 3, 1);   // mlp_cst
  gemm(t3, C3, Pw(90), Pw(91), gc, BS, C3, C3, 1);

  float* fused3 = af((size_t)BS * C3);
  k_fea_fuse<<<BS, 32, 0, stream>>>(gm, ga, gp, gc, Pw(22), Pw(23), fused3, C3);
  float* gcf = af((size_t)BS * C3);
  gemm(fused3, C3, Pw(20), Pw(21), gcf, BS, C3, C3, 1);

  // ---- head + log_softmax ----
  float* h = af((size_t)BS * C2);
  gemm(gcf, C3, Pw(24), Pw(25), h, BS, C2, C3, 1);
  float* logits = af((size_t)BS * NCLS);
  gemm(h, C2, Pw(26), Pw(27), logits, BS, NCLS, C2, 0);
  k_logsoftmax<<<BS, 64, 0, stream>>>(logits, (float*)d_out, NCLS);
}
